// RWKVTimeMixing_65687229825389
// MI455X (gfx1250) — compile-verified, matrix operand order checked by reading, not run on hardware
//
#include <hip/hip_runtime.h>
#include <cstdint>

// Problem constants (match reference)
#define Bb 4
#define Tt 4096
#define Dd 1024
#define Mrows (Bb * Tt)          // 16384 GEMM rows

typedef __attribute__((ext_vector_type(16))) _Float16 v16h;
typedef __attribute__((ext_vector_type(8)))  _Float16 v8h;
typedef __attribute__((ext_vector_type(8)))  float    v8f;

union Frag16 { v16h v16; v8h v8[2]; };

// ---------------- GEMM: out[g,e] = A[g,:] . W[e,:]  (NT, K contiguous both sides)
#define BM 128
#define BN 64
#define BK 32
#define KSTEPS (Dd / BK)
#define PITCH 40   // halves per LDS row (32 + 8 pad)

__device__ inline void mix_store8(_Float16* dst, const float4* xc, const float4* xp,
                                  const float4* tm) {
  float4 c0 = xc[0], c1 = xc[1];
  float4 p0 = xp[0], p1 = xp[1];
  float4 m0 = tm[0], m1 = tm[1];
  v8h h;
  h[0] = (_Float16)(m0.x * c0.x + (1.0f - m0.x) * p0.x);
  h[1] = (_Float16)(m0.y * c0.y + (1.0f - m0.y) * p0.y);
  h[2] = (_Float16)(m0.z * c0.z + (1.0f - m0.z) * p0.z);
  h[3] = (_Float16)(m0.w * c0.w + (1.0f - m0.w) * p0.w);
  h[4] = (_Float16)(m1.x * c1.x + (1.0f - m1.x) * p1.x);
  h[5] = (_Float16)(m1.y * c1.y + (1.0f - m1.y) * p1.y);
  h[6] = (_Float16)(m1.z * c1.z + (1.0f - m1.z) * p1.z);
  h[7] = (_Float16)(m1.w * c1.w + (1.0f - m1.w) * p1.w);
  *(v8h*)dst = h;
}

__device__ inline void cvt_store8(_Float16* dst, const float4* wp) {
  float4 a = wp[0], b = wp[1];
  v8h h;
  h[0] = (_Float16)a.x; h[1] = (_Float16)a.y; h[2] = (_Float16)a.z; h[3] = (_Float16)a.w;
  h[4] = (_Float16)b.x; h[5] = (_Float16)b.y; h[6] = (_Float16)b.z; h[7] = (_Float16)b.w;
  *(v8h*)dst = h;
}

// CDNA5 async DMA: global -> LDS, 16B per lane, tracked by ASYNCcnt.
__device__ inline void async_copy16(unsigned lds_byte_addr, const void* gptr) {
  unsigned long long ga = (unsigned long long)(uintptr_t)gptr;
  asm volatile("global_load_async_to_lds_b128 %0, %1, off"
               :
               : "v"(lds_byte_addr), "v"(ga)
               : "memory");
}
__device__ inline void wait_async0() {
  asm volatile("s_wait_asynccnt 0x0" ::: "memory");
}

// ASRC: 0 = token-shift mix computed on the fly from x/state/tmix (fp32 -> f16, ds_store)
//       1 = A tile async-DMA'd from f16 array a16
// EPI : 0 = plain f32 store, 1 = sigmoid f32 store, 2 = multiply by gate, f32 store
template <int ASRC, int EPI>
__global__ __launch_bounds__(256)
void rwkv_gemm_kernel(const float* __restrict__ x,
                      const float* __restrict__ state,
                      const float* __restrict__ tmix,
                      const _Float16* __restrict__ a16,
                      const _Float16* __restrict__ Wh,   // f16 weights, [N=1024][K=1024]
                      const float* __restrict__ gate,
                      float* __restrict__ out) {
  __shared__ alignas(16) _Float16 As[2][BM * PITCH];
  __shared__ alignas(16) _Float16 Bs[2][BN * PITCH];

  const int tid   = threadIdx.x;
  const int bn    = blockIdx.x;       // N tile (0..15)
  const int bm    = blockIdx.y;       // M tile (0..127)
  const int wid   = tid >> 5;
  const int lane  = tid & 31;
  const int waveM = wid >> 1;         // 0..3, 32 rows each
  const int waveN = wid & 1;          // 0..1, 32 cols each
  const int half  = lane >> 4;        // K-half select per ISA 16-bit A/B layout
  const int lrow  = lane & 15;        // M (or N) within 16

  const int ldrow0 = tid >> 2;        // 64 rows per pass, 4 threads/row
  const int ldcc   = (tid & 3) * 8;   // 8 halves per thread

  v8f acc[2][2] = {};

  // ---- tile loader: fills buffer `buf` with K-slice `kb`
  auto load_tile = [&](int buf, int kb) {
    const int kbase = kb * BK;
    // W tile: 64 rows (n) x 32 k, async DMA (one 16B chunk per thread)
    {
      const int n = ldrow0;
      unsigned lds = (unsigned)(uintptr_t)&Bs[buf][n * PITCH + ldcc];
      async_copy16(lds, Wh + (size_t)(bn * BN + n) * Dd + kbase + ldcc);
    }
    // A tile: 128 rows x 32 k (two passes of 64 rows)
#pragma unroll
    for (int p = 0; p < 2; ++p) {
      const int row = ldrow0 + p * 64;
      const int g   = bm * BM + row;
      if (ASRC == 0) {
        const float4* xc = (const float4*)(x + (size_t)g * Dd + kbase + ldcc);
        const int t = g & (Tt - 1);
        const float* prev = (t == 0)
            ? (state + (size_t)(g >> 12) * Dd + kbase + ldcc)
            : (x + (size_t)(g - 1) * Dd + kbase + ldcc);
        mix_store8(&As[buf][row * PITCH + ldcc], xc, (const float4*)prev,
                   (const float4*)(tmix + kbase + ldcc));
      } else {
        unsigned lds = (unsigned)(uintptr_t)&As[buf][row * PITCH + ldcc];
        async_copy16(lds, a16 + (size_t)g * Dd + kbase + ldcc);
      }
    }
  };

  // prologue: fill buffer 0
  load_tile(0, 0);
  wait_async0();
  __syncthreads();

  for (int kb = 0; kb < KSTEPS; ++kb) {
    const int cur = kb & 1;
    // prefetch next tile into the other buffer (DMA overlaps the WMMAs below)
    if (kb + 1 < KSTEPS) load_tile(cur ^ 1, kb + 1);

    // ---- fragments (A and B share the same 16-bit K-striped layout)
    Frag16 af[2], bf[2];
#pragma unroll
    for (int mi = 0; mi < 2; ++mi) {
      const _Float16* base =
          &As[cur][(waveM * 32 + mi * 16 + lrow) * PITCH + half * 8];
      af[mi].v8[0] = *(const v8h*)base;
      af[mi].v8[1] = *(const v8h*)(base + 16);
    }
#pragma unroll
    for (int ni = 0; ni < 2; ++ni) {
      const _Float16* base =
          &Bs[cur][(waveN * 32 + ni * 16 + lrow) * PITCH + half * 8];
      bf[ni].v8[0] = *(const v8h*)base;
      bf[ni].v8[1] = *(const v8h*)(base + 16);
    }

#pragma unroll
    for (int mi = 0; mi < 2; ++mi)
#pragma unroll
      for (int ni = 0; ni < 2; ++ni)
        acc[mi][ni] = __builtin_amdgcn_wmma_f32_16x16x32_f16(
            false, af[mi].v16, false, bf[ni].v16, (short)0, acc[mi][ni],
            false, false);

    // drain async writes to the next buffer, then rendezvous
    wait_async0();
    __syncthreads();
  }

  // ---- epilogue: C/D layout: VGPR i -> M = i (lanes 0-15) or 8+i (16-31), N = lane&15
  const int mofs = (lane < 16) ? 0 : 8;
#pragma unroll
  for (int mi = 0; mi < 2; ++mi)
#pragma unroll
    for (int ni = 0; ni < 2; ++ni)
#pragma unroll
      for (int i = 0; i < 8; ++i) {
        const int g = bm * BM + waveM * 32 + mi * 16 + mofs + i;
        const int e = bn * BN + waveN * 32 + ni * 16 + lrow;
        const size_t off = (size_t)g * Dd + e;
        float v = acc[mi][ni][i];
        if (EPI == 1) v = 1.0f / (1.0f + expf(-v));
        if (EPI == 2) v *= gate[off];
        out[off] = v;
      }
}

// ---------------- weight fp32 -> f16 pre-conversion ----------------
__global__ __launch_bounds__(256)
void wcvt_kernel(const float* __restrict__ src, _Float16* __restrict__ dst) {
  const size_t i = ((size_t)blockIdx.x * 256 + threadIdx.x) * 8;
  cvt_store8(dst + i, (const float4*)(src + i));
}

// ---------------- Chunked WKV scan ----------------
#define NCH 16
#define CHUNK (Tt / NCH)   // 256

// Pass 1: per-(b,chunk,d) local summary assuming zero incoming state.
__global__ __launch_bounds__(256)
void wkv_local_kernel(const float* __restrict__ kf, const float* __restrict__ vf,
                      const float* __restrict__ w_log,
                      float* __restrict__ sA, float* __restrict__ sB,
                      float* __restrict__ sP) {
  const int idx = blockIdx.x * 256 + threadIdx.x;  // b*NCH*Dd + c*Dd + d
  const int d = idx & (Dd - 1);
  const int c = (idx >> 10) & (NCH - 1);
  const int b = idx >> 14;
  const float W = expf(w_log[d]);                  // decay = exp(-W) per step
  float ah = 0.f, bh = 0.f, pp = -1e30f;
  size_t base = ((size_t)(b * Tt + c * CHUNK)) * Dd + d;
  for (int t = 0; t < CHUNK; ++t) {
    const float kk = kf[base], vv = vf[base];
    const float p  = fmaxf(pp, kk);
    const float e1 = expf(pp - p), e2 = expf(kk - p);
    ah = e1 * ah + e2 * vv;
    bh = e1 * bh + e2;
    pp = p - W;
    base += Dd;
  }
  sA[idx] = ah; sB[idx] = bh; sP[idx] = pp;
}

// Pass 2: per-(b,d) exclusive prefix over 16 chunk summaries -> incoming states.
__global__ __launch_bounds__(256)
void wkv_prefix_kernel(const float* __restrict__ sA, const float* __restrict__ sB,
                       const float* __restrict__ sP, const float* __restrict__ w_log,
                       float* __restrict__ iA, float* __restrict__ iB,
                       float* __restrict__ iP) {
  const int idx = blockIdx.x * 256 + threadIdx.x;  // b*Dd + d
  const int d = idx & (Dd - 1);
  const int b = idx >> 10;
  const float W = expf(w_log[d]);
  float ah = 0.f, bh = 0.f, pp = -1e30f;
  for (int c = 0; c < NCH; ++c) {
    const int s = (b * NCH + c) * Dd + d;
    iA[s] = ah; iB[s] = bh; iP[s] = pp;
    const float dp = pp - (float)CHUNK * W;        // incoming decayed through chunk
    const float lp = sP[s];
    const float p  = fmaxf(dp, lp);
    const float e1 = expf(dp - p), e2 = expf(lp - p);
    ah = e1 * ah + e2 * sA[s];
    bh = e1 * bh + e2 * sB[s];
    pp = p;
  }
}

// Pass 3: replay each chunk with its incoming state, emit wkv (f16 for final GEMM).
__global__ __launch_bounds__(256)
void wkv_replay_kernel(const float* __restrict__ kf, const float* __restrict__ vf,
                       const float* __restrict__ w_log, const float* __restrict__ uu,
                       const float* __restrict__ iA, const float* __restrict__ iB,
                       const float* __restrict__ iP, _Float16* __restrict__ wkv16) {
  const int idx = blockIdx.x * 256 + threadIdx.x;
  const int d = idx & (Dd - 1);
  const int c = (idx >> 10) & (NCH - 1);
  const int b = idx >> 14;
  const float W = expf(w_log[d]);
  const float U = uu[d];
  float ah = iA[idx], bh = iB[idx], pp = iP[idx];
  size_t base = ((size_t)(b * Tt + c * CHUNK)) * Dd + d;
  for (int t = 0; t < CHUNK; ++t) {
    const float kk = kf[base], vv = vf[base];
    // output: (num_hist + e^{u+k} v) / (den_hist + e^{u+k})
    const float ww = U + kk;
    const float qq = fmaxf(pp, ww);
    const float f1 = expf(pp - qq), f2 = expf(ww - qq);
    const float y  = (f1 * ah + f2 * vv) / (f1 * bh + f2 + 1e-38f);
    wkv16[base] = (_Float16)y;
    // state update
    const float p  = fmaxf(pp, kk);
    const float e1 = expf(pp - p), e2 = expf(kk - p);
    ah = e1 * ah + e2 * vv;
    bh = e1 * bh + e2;
    pp = p - W;
    base += Dd;
  }
}

__global__ __launch_bounds__(256)
void copy_state_kernel(const float* __restrict__ x, float* __restrict__ dst) {
  const int idx = blockIdx.x * 256 + threadIdx.x;  // b*Dd + d
  const int d = idx & (Dd - 1);
  const int b = idx >> 10;
  dst[idx] = x[((size_t)b * Tt + (Tt - 1)) * Dd + d];
}

// ---------------- launch ----------------
extern "C" void kernel_launch(void* const* d_in, const int* in_sizes, int n_in,
                              void* d_out, int out_size, void* d_ws, size_t ws_size,
                              hipStream_t stream) {
  (void)in_sizes; (void)n_in; (void)out_size; (void)ws_size;
  const float* x     = (const float*)d_in[0];
  const float* state = (const float*)d_in[1];
  const float* tmr   = (const float*)d_in[2];
  const float* tmk   = (const float*)d_in[3];
  const float* tmv   = (const float*)d_in[4];
  const float* wlog  = (const float*)d_in[5];
  const float* uu    = (const float*)d_in[6];
  const float* W_r   = (const float*)d_in[7];
  const float* W_k   = (const float*)d_in[8];
  const float* W_v   = (const float*)d_in[9];
  const float* W_o   = (const float*)d_in[10];
  float* out = (float*)d_out;

  const size_t MD = (size_t)Mrows * Dd;          // 16.7M
  const size_t WD = (size_t)Dd * Dd;             // 1M
  char* ws = (char*)d_ws;
  float*    r_f   = (float*)ws;                  // MD f32
  float*    k_f   = r_f + MD;                    // MD f32
  float*    v_f   = k_f + MD;                    // MD f32
  _Float16* wkv16 = (_Float16*)(v_f + MD);       // MD f16
  _Float16* Wh_r  = wkv16 + MD;                  // 4 x WD f16
  _Float16* Wh_k  = Wh_r + WD;
  _Float16* Wh_v  = Wh_k + WD;
  _Float16* Wh_o  = Wh_v + WD;
  float*    sA    = (float*)(Wh_o + WD);         // 6 x B*NCH*Dd f32
  const size_t SN = (size_t)Bb * NCH * Dd;
  float* sB = sA + SN;
  float* sP = sB + SN;
  float* iA = sP + SN;
  float* iB = iA + SN;
  float* iP = iB + SN;

  // weights fp32 -> f16 (once per launch; tiny)
  const int wblocks = (int)(WD / 8 / 256);       // 512
  wcvt_kernel<<<wblocks, 256, 0, stream>>>(W_r, Wh_r);
  wcvt_kernel<<<wblocks, 256, 0, stream>>>(W_k, Wh_k);
  wcvt_kernel<<<wblocks, 256, 0, stream>>>(W_v, Wh_v);
  wcvt_kernel<<<wblocks, 256, 0, stream>>>(W_o, Wh_o);

  dim3 ggrid(Dd / BN, Mrows / BM);               // (16, 128)
  dim3 gblk(256);

  // r = sigmoid(xr @ W_r^T), k = xk @ W_k^T, v = xv @ W_v^T  (mix fused in loader)
  rwkv_gemm_kernel<0, 1><<<ggrid, gblk, 0, stream>>>(x, state, tmr, nullptr, Wh_r, nullptr, r_f);
  rwkv_gemm_kernel<0, 0><<<ggrid, gblk, 0, stream>>>(x, state, tmk, nullptr, Wh_k, nullptr, k_f);
  rwkv_gemm_kernel<0, 0><<<ggrid, gblk, 0, stream>>>(x, state, tmv, nullptr, Wh_v, nullptr, v_f);

  // chunked parallel WKV scan
  const int nScan = Bb * NCH * Dd;               // 65536
  wkv_local_kernel <<<nScan / 256, 256, 0, stream>>>(k_f, v_f, wlog, sA, sB, sP);
  wkv_prefix_kernel<<<(Bb * Dd) / 256, 256, 0, stream>>>(sA, sB, sP, wlog, iA, iB, iP);
  wkv_replay_kernel<<<nScan / 256, 256, 0, stream>>>(k_f, v_f, wlog, uu, iA, iB, iP, wkv16);

  // out = r * (wkv @ W_o^T)
  rwkv_gemm_kernel<1, 2><<<ggrid, gblk, 0, stream>>>(x, state, nullptr, wkv16, Wh_o, r_f, out);

  // new_state = x[:, -1, :]
  copy_state_kernel<<<(Bb * Dd) / 256, 256, 0, stream>>>(x, out + MD);
}
